// QuantizationLayer_20229295964690
// MI455X (gfx1250) — matrix operand order repaired and checked by reading, not compiled, not run on hardware
//
#include <hip/hip_runtime.h>
#include <hip/hip_bf16.h>

// ---- problem constants (match reference) ----
#define NEVT   2000000
#define Cd     9
#define Hd     180
#define Wd     240
#define Bd     4
#define OUTR   256
#define NEG    0.1f
#define VOXN   (Bd * 2 * Cd * Hd * Wd)      // 3,110,400 floats (12.4 MB, L2-resident)
#define HIDP   128                          // hidden 100 padded to 128 for WMMA
#define BSTRIDE 136                         // LDS row stride (halves): conflict-free b128 loads

typedef __attribute__((ext_vector_type(16))) _Float16 v16h;
typedef __attribute__((ext_vector_type(8)))  _Float16 v8h;
typedef __attribute__((ext_vector_type(8)))  float    v8f;

static __device__ __forceinline__ v16h splat16(float f) {
    v16h v;
    _Float16 h = (_Float16)f;
#pragma unroll
    for (int i = 0; i < 16; ++i) v[i] = h;
    return v;
}

// ---------------- kernel 0: zero vox + tmax scratch ----------------
__global__ void k_zero(float* ws, int n) {
    int i = blockIdx.x * blockDim.x + threadIdx.x;
    int stride = gridDim.x * blockDim.x;
    for (; i < n; i += stride) ws[i] = 0.0f;
}

// ---------------- kernel 1: per-batch max(t) (t > 0 -> uint bits order-preserving) ----------------
__global__ void k_segmax(const float* __restrict__ ev, unsigned* __restrict__ tmax) {
    int i = blockIdx.x * blockDim.x + threadIdx.x;
    int stride = gridDim.x * blockDim.x;
    for (; i < NEVT; i += stride) {
        float t  = ev[i * 5 + 2];
        int   bi = (int)ev[i * 5 + 4];
        atomicMax(&tmax[bi], __float_as_uint(t));
    }
}

// ---------------- kernel 2: MLP voxelization via WMMA f32_16x16x32_f16 ----------------
__global__ void __launch_bounds__(256)
k_voxel(const float* __restrict__ ev,
        const float* __restrict__ gW1, const float* __restrict__ gb1,
        const float* __restrict__ gW2, const float* __restrict__ gb2,
        const float* __restrict__ gW3, const float* __restrict__ gb3,
        const unsigned* __restrict__ tmaxb,
        float* __restrict__ vox)
{
    // W2 staged transposed: sB[n*BSTRIDE + k] = W2[k][n] (f16, zero-padded to 128x128)
    __shared__ __align__(16) _Float16 sB[HIDP * BSTRIDE];
    __shared__ __align__(16) _Float16 sW1h[HIDP], sb1h[HIDP];
    __shared__ float sb2[HIDP], sW3[HIDP];

    const int tid = threadIdx.x;
    for (int i = tid; i < HIDP * HIDP; i += 256) {
        int n = i >> 7, k = i & 127;
        float v = (k < 100 && n < 100) ? gW2[k * 100 + n] : 0.0f;
        sB[n * BSTRIDE + k] = (_Float16)v;
    }
    if (tid < HIDP) {
        sW1h[tid] = (_Float16)((tid < 100) ? gW1[tid] : 0.0f);
        sb1h[tid] = (_Float16)((tid < 100) ? gb1[tid] : 0.0f);
        sb2[tid]  = (tid < 100) ? gb2[tid] : 0.0f;
        sW3[tid]  = (tid < 100) ? gW3[tid] : 0.0f;
    }
    __syncthreads();

    const float b3 = gb3[0];
    const int lane  = tid & 31;
    const int hf    = lane >> 4;          // lane half (0/1): K split per ISA A layout
    const int lrow  = lane & 15;          // M row / N column within 16
    const int gwave = blockIdx.x * 8 + (tid >> 5);
    const int nwaves = gridDim.x * 8;
    const int NT = NEVT / 16;             // 125,000 tiles of 16 events

    const v16h vzero = {};
    const v16h vneg  = splat16(NEG);

    for (int tile = gwave; tile < NT; tile += nwaves) {
        int  e     = tile * 16 + lrow;
        bool valid = (e < NEVT);
        int  eb    = valid ? e : 0;
        float x  = ev[eb * 5 + 0];
        float y  = ev[eb * 5 + 1];
        float t  = ev[eb * 5 + 2];
        float p  = ev[eb * 5 + 3];
        float bb = ev[eb * 5 + 4];
        int   bi = (int)bb;
        float tm = __uint_as_float(tmaxb[bi]);
        float tn = t / tm;                                 // normalized t
        int idx0 = (int)x + Wd * (int)y + Wd * Hd * Cd * (int)p + Wd * Hd * Cd * 2 * bi;

        // Bin-invariant row broadcasts (D layout: VGPR r on half hf -> row m = r + 8*hf)
        float tnm[8]; int idxm[8]; int vmr[8];
#pragma unroll
        for (int r = 0; r < 8; ++r) {
            int m = r + (hf << 3);
            tnm[r]  = __shfl(tn, m, 32);
            idxm[r] = __shfl(idx0, m, 32);
            vmr[r]  = __shfl(valid ? 1 : 0, m, 32);
        }

        for (int bin = 0; bin < Cd; ++bin) {
            // A fragments rebuilt per bin, fully packed f16.
            // A layout per ISA (16-bit 16x32): for (ks,hf) the K indices are two contiguous
            // 8-half runs (k = ks*32 + hf*8 + {0..7} and +16) -> 2x 16B broadcast ds_load_b128.
            // W1/b1 zero-padded for k>=100 -> lrelu(0)=0, no masking needed.
            v16h a[4];
            {
                v16h ts16 = splat16(tn - (float)bin * 0.125f);
#pragma unroll
                for (int ks = 0; ks < 4; ++ks) {
                    int k0 = ks * 32 + (hf << 3);
                    v8h w1lo = *(const v8h*)&sW1h[k0];
                    v8h w1hi = *(const v8h*)&sW1h[k0 + 16];
                    v8h b1lo = *(const v8h*)&sb1h[k0];
                    v8h b1hi = *(const v8h*)&sb1h[k0 + 16];
                    v16h w1f = __builtin_shufflevector(w1lo, w1hi,
                                0,1,2,3,4,5,6,7,8,9,10,11,12,13,14,15);
                    v16h b1f = __builtin_shufflevector(b1lo, b1hi,
                                0,1,2,3,4,5,6,7,8,9,10,11,12,13,14,15);
                    v16h pp  = w1f * ts16 + b1f;                       // v_pk_fma_f16
                    v16h pos = __builtin_elementwise_max(pp, vzero);
                    v16h neg = __builtin_elementwise_min(pp, vzero);
                    a[ks] = pos + neg * vneg;                          // leaky ReLU
                }
            }

            float partial[8];
#pragma unroll
            for (int r = 0; r < 8; ++r) partial[r] = 0.0f;

            for (int nt = 0; nt < 8; ++nt) {               // 8 N-tiles of 16 (hidden padded to 128)
                v8f acc = {};
                int n = nt * 16 + lrow;
#pragma unroll
                for (int ks = 0; ks < 4; ++ks) {
                    // B fragment: column n, K rows ks*32 + hf*8 + {0..7} and +16 -> 2x ds_load_b128
                    const _Float16* bp = &sB[n * BSTRIDE + ks * 32 + (hf << 3)];
                    v8h blo = *(const v8h*)(bp);
                    v8h bhi = *(const v8h*)(bp + 16);
                    v16h bf = __builtin_shufflevector(blo, bhi,
                               0,1,2,3,4,5,6,7,8,9,10,11,12,13,14,15);
                    acc = __builtin_amdgcn_wmma_f32_16x16x32_f16(
                              false, a[ks], false, bf, (short)0, acc, false, false);
                }
                // fold bias + leaky_relu + dot with W3 (b2/W3 zero-padded for n>=100)
                float bn = sb2[n], wn = sW3[n];
#pragma unroll
                for (int r = 0; r < 8; ++r) {
                    float h2 = acc[r] + bn;
                    h2 = (h2 > 0.0f) ? h2 : NEG * h2;
                    partial[r] += h2 * wn;
                }
            }

            // reduce across the 16 lanes of each half (wave32 butterfly)
#pragma unroll
            for (int r = 0; r < 8; ++r)
#pragma unroll
                for (int off = 1; off < 16; off <<= 1)
                    partial[r] += __shfl_xor(partial[r], off, 32);

#pragma unroll
            for (int r = 0; r < 8; ++r) {
                float val = tnm[r] * (partial[r] + b3);
                if (lrow == 0 && vmr[r])
                    atomicAdd(&vox[idxm[r] + Wd * Hd * bin], val);   // L2-resident scatter
            }
        }
    }
}

// ---------------- kernel 3: symmetric pad (180->240) + nearest resize (240->256) ----------------
__global__ void k_resize(const float* __restrict__ vox, float* __restrict__ out) {
    const int total = Bd * 2 * Cd * OUTR * OUTR;
    int i = blockIdx.x * blockDim.x + threadIdx.x;
    if (i >= total) return;
    int ox = i & 255;
    int oy = (i >> 8) & 255;
    int c  = (i >> 16) % (2 * Cd);
    int b  = i / (OUTR * OUTR * 2 * Cd);
    int ix  = (ox * Wd) >> 8;        // (ox*240)//256
    int iyp = (oy * Wd) >> 8;        // padded-row index in [0,240)
    int yy  = iyp - 30;              // remove symmetric pad
    float v = 0.0f;
    if (yy >= 0 && yy < Hd) {
        int pp = c / Cd, bin = c % Cd;
        v = vox[ix + Wd * (yy + Hd * (bin + Cd * (pp + 2 * b)))];
    }
    out[i] = v;
}

extern "C" void kernel_launch(void* const* d_in, const int* in_sizes, int n_in,
                              void* d_out, int out_size, void* d_ws, size_t ws_size,
                              hipStream_t stream) {
    const float* ev  = (const float*)d_in[0];
    const float* W1  = (const float*)d_in[1];
    const float* b1  = (const float*)d_in[2];
    const float* W2  = (const float*)d_in[3];
    const float* b2  = (const float*)d_in[4];
    const float* W3  = (const float*)d_in[5];
    const float* b3  = (const float*)d_in[6];
    float*       out = (float*)d_out;

    float*    vox  = (float*)d_ws;
    unsigned* tmax = (unsigned*)((char*)d_ws + (size_t)VOXN * sizeof(float));

    // zero vox + 4 tmax words (uint 0 == float 0 bits)
    k_zero<<<2048, 256, 0, stream>>>(vox, VOXN + 4);
    k_segmax<<<2048, 256, 0, stream>>>(ev, tmax);
    k_voxel<<<1024, 256, 0, stream>>>(ev, W1, b1, W2, b2, W3, b3, tmax, vox);
    const int total = Bd * 2 * Cd * OUTR * OUTR;
    k_resize<<<(total + 255) / 256, 256, 0, stream>>>(vox, out);
}
